// CrossAttention_34737695490205
// MI455X (gfx1250) — matrix operand order repaired and checked by reading, not compiled, not run on hardware
//
#include <hip/hip_runtime.h>
#include <hip/hip_bf16.h>

// CDNA5 / gfx1250, wave32. All matmuls via v_wmma_f32_16x16x32_bf16.
// K staging uses GLOBAL_LOAD_ASYNC_TO_LDS_B128 (ASYNCcnt) with double buffering
// when the toolchain exposes the builtin.

typedef __bf16 bf16_t;
typedef __attribute__((ext_vector_type(16))) __bf16 v16bf;
typedef __attribute__((ext_vector_type(8)))  float  v8f;
typedef __attribute__((ext_vector_type(4)))  int    v4i;

#define DEV __device__ __forceinline__

#if defined(__AMDGCN__) && __has_builtin(__builtin_amdgcn_global_load_async_to_lds_b128)
#define HAS_GLDS_ASYNC 1
#else
#define HAS_GLDS_ASYNC 0
#endif

// ---------------- helpers ----------------

DEV unsigned short f2bf(float f) {
  bf16_t h = (bf16_t)f;
  return __builtin_bit_cast(unsigned short, h);
}
DEV float bf2f(unsigned short u) {
  return (float)__builtin_bit_cast(bf16_t, u);
}

DEV v8f wmma_bf16(v16bf a, v16bf b, v8f c) {
  // D = A(16x32) * B(32x16) + C(16x16 f32)
  return __builtin_amdgcn_wmma_f32_16x16x32_bf16(
      false, a, false, b, (short)0, c, false, false);
}

// 16-byte global->LDS copy. Async (no VGPR round trip) when available.
// Toolchain prototype: (v4i* global_src, v4i* lds_dst, imm offset, imm cpol).
DEV void copy16_g2l(const unsigned short* g, unsigned short* l) {
#if HAS_GLDS_ASYNC
  __builtin_amdgcn_global_load_async_to_lds_b128((v4i*)g, (v4i*)l, 0, 0);
#else
  const unsigned int* gs = (const unsigned int*)g;
  unsigned int* ls = (unsigned int*)l;
  ls[0] = gs[0]; ls[1] = gs[1]; ls[2] = gs[2]; ls[3] = gs[3];
#endif
}

template <int N> DEV void wait_async() {
#if HAS_GLDS_ASYNC
#if __has_builtin(__builtin_amdgcn_s_wait_asynccnt)
  __builtin_amdgcn_s_wait_asynccnt(N);
#else
  asm volatile("s_wait_asynccnt %0" :: "i"(N) : "memory");
#endif
#endif
}

// A-operand fragment: pointer to THIS lane's row (32 contiguous bf16, dword
// aligned). Contiguous K order (matches frag_b):
//   lanes 0-15 : dword j in 0..7 -> K = 2j, 2j+1
//   lanes16-31 : dword j in 0..7 -> K = 16+2j, 17+2j
DEV v16bf frag_a(const unsigned short* row, int half) {
  const unsigned int* p = (const unsigned int*)row;
  union { v16bf v; unsigned int u[8]; } r;
  int o = half * 8;
#pragma unroll
  for (int j = 0; j < 8; ++j) r.u[j] = p[o + j];
  return r.v;
}

// B-operand fragment: pointer to THIS lane's column data contiguous over K
// (32 bf16, dword aligned). Same contiguous K order as frag_a:
//   lanes 0-15: vgpr j -> K = 2j, 2j+1 ; lanes 16-31: K = 16+2j, 17+2j
DEV v16bf frag_b(const unsigned short* col, int half) {
  const unsigned int* p = (const unsigned int*)col;
  union { v16bf v; unsigned int u[8]; } r;
  int o = half * 8;
#pragma unroll
  for (int j = 0; j < 8; ++j) r.u[j] = p[o + j];
  return r.v;
}

// ---------------- kernel 1/4: generic M x 256 x 256 GEMM ----------------
// X: (M,256) f32 row-major. W: (256,256) f32 row-major (K-major).
// mode 0: out bf16 at [b][h][s][e]   (b=row>>11, s=row&2047, h=n>>5, e=n&31)
// mode 1: out bf16 at [b][h][e][s]   (transposed, for V)
// mode 2: out f32 row-major with bias + leaky-relu (final projection)
// scale multiplies (X*W + bias) before bf16 store (q is pre-scaled by 2).
__global__ __launch_bounds__(256) void gemm256_kernel(
    const float* __restrict__ X, const float* __restrict__ W,
    const float* __restrict__ bias, void* __restrict__ out,
    int mode, float scale) {
  __shared__ __align__(16) unsigned short As[64 * 258];   // 64 rows x 256 K
  __shared__ __align__(16) unsigned short Ws[256 * 34];   // [n][k-slice of 32]

  int tid = threadIdx.x;
  int lane = tid & 31, wave = tid >> 5;
  int half = lane >> 4, l16 = lane & 15;
  int row0 = blockIdx.x * 64;

  // Stage full A tile (64 x 256), f32 -> bf16, coalesced.
  for (int idx = tid; idx < 64 * 256; idx += 256) {
    int r = idx >> 8, k = idx & 255;
    As[r * 258 + k] = f2bf(X[(row0 + r) * 256 + k]);
  }

  int mt = wave & 3;        // m-tile: rows mt*16 .. +16
  int nh = wave >> 2;       // n-half: columns nh*128 .. +128
  v8f acc[8];
#pragma unroll
  for (int i = 0; i < 8; ++i)
#pragma unroll
    for (int j = 0; j < 8; ++j) acc[i][j] = 0.0f;

  for (int ks = 0; ks < 8; ++ks) {
    __syncthreads();
    // Stage W k-slice transposed: Ws[n][k] = W[ks*32+k][n]
    for (int idx = tid; idx < 32 * 256; idx += 256) {
      int k = idx >> 8, n = idx & 255;
      Ws[n * 34 + k] = f2bf(W[(ks * 32 + k) * 256 + n]);
    }
    __syncthreads();
    v16bf a = frag_a(As + (mt * 16 + l16) * 258 + ks * 32, half);
#pragma unroll
    for (int nt = 0; nt < 8; ++nt) {
      int n = nh * 128 + nt * 16 + l16;
      v16bf b = frag_b(Ws + n * 34, half);
      acc[nt] = wmma_bf16(a, b, acc[nt]);
    }
  }

  // Epilogue. C layout: vgpr r -> M = r + 8*half, N = l16.
#pragma unroll
  for (int nt = 0; nt < 8; ++nt) {
    int n = nh * 128 + nt * 16 + l16;
    float bi = bias[n];
#pragma unroll
    for (int r = 0; r < 8; ++r) {
      int row = row0 + mt * 16 + half * 8 + r;
      float val = acc[nt][r] + bi;
      if (mode == 2) {
        val = val >= 0.0f ? val : 0.01f * val;
        ((float*)out)[row * 256 + n] = val;
      } else {
        val *= scale;
        int b = row >> 11, s = row & 2047;
        int h = n >> 5, e = n & 31;
        long off;
        if (mode == 0) off = ((long)(b * 8 + h) * 2048 + s) * 32 + e;
        else           off = ((long)(b * 8 + h) * 32 + e) * 2048 + s;
        ((unsigned short*)out)[off] = f2bf(val);
      }
    }
  }
}

// ---------------- kernel 2: key squared norms ----------------
__global__ __launch_bounds__(256) void k2_kernel(
    const unsigned short* __restrict__ kbf, float* __restrict__ k2) {
  int i = blockIdx.x * 256 + threadIdx.x;        // exactly B*H*SK = 32768
  const unsigned short* p = kbf + i * 32;
  float s = 0.0f;
#pragma unroll
  for (int e = 0; e < 32; ++e) { float v = bf2f(p[e]); s += v * v; }
  k2[i] = s;
}

// ---------------- kernel 3: online softmax stats over keys ----------------
// score(q,k) = 2*q.k - k2[k]  (q2 cancels in the softmax). q is pre-scaled x2.
// K chunks (128 keys x 32 e) stream into LDS via async global->LDS with double
// buffering; stride 40 ushorts keeps every b128 destination 16B-aligned.
__global__ __launch_bounds__(256) void stats_kernel(
    const unsigned short* __restrict__ qbf, const unsigned short* __restrict__ kbf,
    const float* __restrict__ k2, float* __restrict__ maxs, float* __restrict__ sums) {
  __shared__ __align__(16) unsigned short Ks[2][128 * 40];
  __shared__ float k2s[2][128];

  int bh = blockIdx.x >> 4;          // 16 q-blocks per (b,h)
  int qblk = blockIdx.x & 15;
  int tid = threadIdx.x, lane = tid & 31, wave = tid >> 5;
  int half = lane >> 4, l16 = lane & 15;

  const unsigned short* qb = qbf + bh * 2048 * 32;
  const unsigned short* kb = kbf + bh * 2048 * 32;
  const float* k2b = k2 + bh * 2048;

  int q0 = qblk * 128 + wave * 16;
  v16bf aq = frag_a(qb + (q0 + l16) * 32, half);   // Q tile fixed for this wave

  float rmax[8], rsum[8];
#pragma unroll
  for (int r = 0; r < 8; ++r) { rmax[r] = -3.0e38f; rsum[r] = 0.0f; }

  // Stage chunk c into buffer bufi: 512 x 16B transfers, 2 per thread.
  auto stage = [&](int c, int bufi) {
    const unsigned short* src = kb + c * 128 * 32;
#pragma unroll
    for (int i = 0; i < 2; ++i) {
      int idx = tid + i * 256;
      int row = idx >> 2, seg = idx & 3;
      copy16_g2l(src + row * 32 + seg * 8, &Ks[bufi][row * 40 + seg * 8]);
    }
    if (tid < 128) k2s[bufi][tid] = k2b[c * 128 + tid];
  };

  stage(0, 0);
  for (int kc = 0; kc < 16; ++kc) {
    int bufi = kc & 1;
    if (kc < 15) { stage(kc + 1, bufi ^ 1); wait_async<2>(); }
    else         { wait_async<0>(); }
    __syncthreads();                       // chunk kc staged & visible

#pragma unroll 1
    for (int t = 0; t < 8; ++t) {
      v16bf bk = frag_b(&Ks[bufi][(t * 16 + l16) * 40], half); // B = K^T
      v8f z;
#pragma unroll
      for (int j = 0; j < 8; ++j) z[j] = 0.0f;
      v8f c = wmma_bf16(aq, bk, z);                            // c = 2*q.k
      float k2l = k2s[bufi][t * 16 + l16];
#pragma unroll
      for (int r = 0; r < 8; ++r) {
        float sc = c[r] - k2l;
        float cm = sc;                                   // row-max over 16 lanes
        cm = fmaxf(cm, __shfl_xor(cm, 1, 32));
        cm = fmaxf(cm, __shfl_xor(cm, 2, 32));
        cm = fmaxf(cm, __shfl_xor(cm, 4, 32));
        cm = fmaxf(cm, __shfl_xor(cm, 8, 32));
        float nm = fmaxf(rmax[r], cm);
        float ev = __expf(sc - nm);
        ev += __shfl_xor(ev, 1, 32);
        ev += __shfl_xor(ev, 2, 32);
        ev += __shfl_xor(ev, 4, 32);
        ev += __shfl_xor(ev, 8, 32);
        rsum[r] = rsum[r] * __expf(rmax[r] - nm) + ev;
        rmax[r] = nm;
      }
    }
    __syncthreads();                       // all waves done with buffer bufi
  }
  if (l16 == 0) {
#pragma unroll
    for (int r = 0; r < 8; ++r) {
      int q = q0 + half * 8 + r;                   // C row M = r + 8*half
      maxs[bh * 2048 + q] = rmax[r];
      sums[bh * 2048 + q] = rsum[r];
    }
  }
}

// ---------------- kernel 4: attn[k,e] = sum_q w[q,k] * v[q,e] ----------------
// Per wave: one 16-key tile. scores^T via WMMA(K_tile, Q^T) -> (k,q) oriented,
// exp + normalize, bounce through per-wave LDS into A-fragment layout, then
// accumulate against v^T with two WMMAs (e tiles 0..15, 16..31).
__global__ __launch_bounds__(256) void attn_accum_kernel(
    const unsigned short* __restrict__ qbf, const unsigned short* __restrict__ kbf,
    const unsigned short* __restrict__ vbfT, const float* __restrict__ k2,
    const float* __restrict__ maxs, const float* __restrict__ sums,
    float* __restrict__ attn) {
  __shared__ __align__(16) unsigned short wlds[8 * 16 * 34];

  int bh = blockIdx.x >> 4;          // 16 k-blocks per (b,h)
  int kblk = blockIdx.x & 15;
  int b = bh >> 3, h = bh & 7;
  int tid = threadIdx.x, lane = tid & 31, wave = tid >> 5;
  int half = lane >> 4, l16 = lane & 15;
  unsigned short* wt = wlds + wave * 16 * 34;      // private 16x32 w^T tile

  const unsigned short* qb = qbf + bh * 2048 * 32;
  const unsigned short* kb = kbf + bh * 2048 * 32;
  const unsigned short* vb = vbfT + bh * 32 * 2048;

  int k0 = kblk * 128 + wave * 16;
  v16bf ak = frag_a(kb + (k0 + l16) * 32, half);   // K tile fixed per wave
  float k2r[8];
#pragma unroll
  for (int r = 0; r < 8; ++r) k2r[r] = k2[bh * 2048 + k0 + half * 8 + r];

  v8f acc0, acc1;
#pragma unroll
  for (int j = 0; j < 8; ++j) { acc0[j] = 0.0f; acc1[j] = 0.0f; }

  for (int qc = 0; qc < 64; ++qc) {                // 32 queries per iter
    if (qc + 1 < 64) {                             // global_prefetch_b8 path
      __builtin_prefetch(qb + (qc * 32 + 32 + l16) * 32, 0, 3);
      __builtin_prefetch(vb + l16 * 2048 + (qc + 1) * 32, 0, 3);
      __builtin_prefetch(vb + (16 + l16) * 2048 + (qc + 1) * 32, 0, 3);
    }
#pragma unroll
    for (int sub = 0; sub < 2; ++sub) {
      int q = qc * 32 + sub * 16 + l16;
      v16bf bq = frag_b(qb + q * 32, half);        // B = Q^T (col = q)
      v8f z;
#pragma unroll
      for (int j = 0; j < 8; ++j) z[j] = 0.0f;
      v8f c = wmma_bf16(ak, bq, z);                // (k,q) scores^T = 2*q.k
      float mx = maxs[bh * 2048 + q];
      float is = 1.0f / sums[bh * 2048 + q];
#pragma unroll
      for (int r = 0; r < 8; ++r) {                // row = k-in-tile = r+8*half
        float w = __expf(c[r] - k2r[r] - mx) * is;
        wt[(half * 8 + r) * 34 + sub * 16 + l16] = f2bf(w);
      }
    }
    // DS ops are in-order within a wave -> no barrier needed (per-wave region)
    v16bf aw  = frag_a(wt + l16 * 34, half);                   // A = w^T 16x32q
    v16bf bv0 = frag_b(vb + l16 * 2048 + qc * 32, half);       // B = v, e 0..15
    v16bf bv1 = frag_b(vb + (16 + l16) * 2048 + qc * 32, half);// e 16..31
    acc0 = wmma_bf16(aw, bv0, acc0);
    acc1 = wmma_bf16(aw, bv1, acc1);
  }

#pragma unroll
  for (int r = 0; r < 8; ++r) {
    int k = k0 + half * 8 + r;
    long base = ((long)(b * 2048 + k) * 8 + h) * 32;   // attn[b][k][h][e]
    attn[base + l16]      = acc0[r];
    attn[base + 16 + l16] = acc1[r];
  }
}

// ---------------- launch ----------------

extern "C" void kernel_launch(void* const* d_in, const int* in_sizes, int n_in,
                              void* d_out, int out_size, void* d_ws, size_t ws_size,
                              hipStream_t stream) {
  (void)in_sizes; (void)n_in; (void)out_size; (void)ws_size;
  const float* x  = (const float*)d_in[0];
  const float* t  = (const float*)d_in[1];
  const float* Wk = (const float*)d_in[2];
  const float* bk = (const float*)d_in[3];
  const float* Wq = (const float*)d_in[4];
  const float* bq = (const float*)d_in[5];
  const float* Wv = (const float*)d_in[6];
  const float* bv = (const float*)d_in[7];
  const float* Wo = (const float*)d_in[8];
  const float* bo = (const float*)d_in[9];

  char* ws = (char*)d_ws;
  unsigned short* qbf = (unsigned short*)(ws);                     // 2 MB  [b,h,q,e] bf16 (x2)
  unsigned short* kbf = (unsigned short*)(ws + (2u << 20));        // 2 MB  [b,h,k,e] bf16
  unsigned short* vbf = (unsigned short*)(ws + (4u << 20));        // 2 MB  [b,h,e,q] bf16
  float* k2   = (float*)(ws + (6u << 20));                         // 128 KB
  float* mx   = (float*)(ws + (6u << 20) + (128u << 10));          // 128 KB
  float* sm   = (float*)(ws + (6u << 20) + (256u << 10));          // 128 KB
  float* attn = (float*)(ws + (8u << 20));                         // 4 MB  [b,k,h,e] f32

  dim3 blk(256);
  // QKV projections (q pre-scaled by 2 so scores = WMMA output directly)
  gemm256_kernel<<<64, blk, 0, stream>>>(t, Wq, bq, (void*)qbf, 0, 2.0f);
  gemm256_kernel<<<64, blk, 0, stream>>>(x, Wk, bk, (void*)kbf, 0, 1.0f);
  gemm256_kernel<<<64, blk, 0, stream>>>(t, Wv, bv, (void*)vbf, 1, 1.0f);
  // key norms
  k2_kernel<<<128, blk, 0, stream>>>(kbf, k2);
  // softmax stats (flash-style, no materialized attention matrix)
  stats_kernel<<<256, blk, 0, stream>>>(qbf, kbf, k2, mx, sm);
  // transposed-contraction accumulation into attn[b,k,h,e]
  attn_accum_kernel<<<256, blk, 0, stream>>>(qbf, kbf, vbf, k2, mx, sm, attn);
  // output projection + bias + leaky relu -> d_out (f32, [b,k,o])
  gemm256_kernel<<<64, blk, 0, stream>>>(attn, Wo, bo, d_out, 2, 1.0f);
}
